// IDMRFLoss_12618613915712
// MI455X (gfx1250) — matrix operand order repaired and checked by reading, not compiled, not run on hardware
//
#include <hip/hip_runtime.h>

typedef __attribute__((ext_vector_type(16))) _Float16 v16h;
typedef __attribute__((ext_vector_type(8)))  _Float16 v8h;
typedef __attribute__((ext_vector_type(8)))  float    v8f;

union Frag16 { v16h v; v8h h[2]; };

#define EPS_DIV 1e-5f
#define EPS_LOG 1e-5f

static constexpr int NB = 4;
static constexpr int CA = 256, PA = 4096;   // relu3_2: [4,256,64,64], P = 64*64
static constexpr int CB = 512, PB = 1024;   // relu4_2: [4,512,32,32], P = 32*32

// ---------------------------------------------------------------------------
// Preprocess: meanT over channels, subtract, L2-normalize along C, write f16
// operands transposed to patch-major [B][P][C] so WMMA fragments are
// contiguous 16B per-lane loads.
// ---------------------------------------------------------------------------
template<int C, int P>
__global__ void prep_kernel(const float* __restrict__ gen, const float* __restrict__ tar,
                            _Float16* __restrict__ Gn, _Float16* __restrict__ Tn) {
    const int b = blockIdx.y;
    const int q = blockIdx.x * blockDim.x + threadIdx.x;
    const float* tb = tar + ((size_t)b * C) * P + q;
    const float* gb = gen + ((size_t)b * C) * P + q;

    float s = 0.f;
    #pragma unroll 4
    for (int c = 0; c < C; ++c) s += tb[(size_t)c * P];
    const float mean = s / (float)C;

    float st = 0.f, sg = 0.f;
    #pragma unroll 4
    for (int c = 0; c < C; ++c) {
        float tv = tb[(size_t)c * P] - mean;
        float gv = gb[(size_t)c * P] - mean;
        st += tv * tv;
        sg += gv * gv;
    }
    const float rt = rsqrtf(st);
    const float rg = rsqrtf(sg);

    _Float16* trow = Tn + ((size_t)b * P + q) * C;
    _Float16* grow = Gn + ((size_t)b * P + q) * C;
    #pragma unroll 4
    for (int c = 0; c < C; ++c) {
        trow[c] = (_Float16)((tb[(size_t)c * P] - mean) * rt);
        grow[c] = (_Float16)((gb[(size_t)c * P] - mean) * rg);
    }
}

// ---------------------------------------------------------------------------
// One 16x16 cosine tile: A rows = tar patches p (from Tbase row of this lane),
// B cols = gen pixels q (from Gbase row of this lane). K loop over channels.
// Per-lane fragment layout per CDNA5 ISA 7.12.2:
//   A (16x32 f16): lane<16 holds K 0..7 & 16..23; lane>=16 holds K 8..15 & 24..31
//   B (32x16 f16): lane<16 holds K 0..7 & 16..23; lane>=16 holds K 8..15 & 24..31
// ---------------------------------------------------------------------------
template<int C>
__device__ inline v8f cosine_tile(const _Float16* __restrict__ Tbase,
                                  const _Float16* __restrict__ Gbase, int hi) {
    v8f acc = {};
    #pragma unroll
    for (int kc = 0; kc < C; kc += 32) {
        Frag16 a, bb;
        a.h[0]  = *(const v8h*)(Tbase + kc + 8 * hi);
        a.h[1]  = *(const v8h*)(Tbase + kc + 16 + 8 * hi);
        bb.h[0] = *(const v8h*)(Gbase + kc + 8 * hi);
        bb.h[1] = *(const v8h*)(Gbase + kc + 16 + 8 * hi);
        acc = __builtin_amdgcn_wmma_f32_16x16x32_f16(
            /*neg_a=*/false, a.v, /*neg_b=*/false, bb.v,
            /*c_mod=*/(short)0, acc, /*reuse_a=*/false, /*reuse_b=*/false);
    }
    return acc;
}

// ---------------------------------------------------------------------------
// Pass 1: per-column (per gen pixel q) minimum of cdist over all tar patches p.
// Block owns 64 columns; 8 waves = 4 q-subtiles x 2 p-phases; loop p.
// D layout: lane holds column n = lane&15, rows m = (lane>>4)*8 + r.
// ---------------------------------------------------------------------------
template<int C, int P>
__global__ void colmin_kernel(const _Float16* __restrict__ Tn, const _Float16* __restrict__ Gn,
                              float* __restrict__ minc) {
    __shared__ unsigned smin[64];
    const int tid = threadIdx.x;
    const int wave = tid >> 5, lane = tid & 31, lo = lane & 15, hi = lane >> 4;
    const int b = blockIdx.y;
    const int q0 = blockIdx.x * 64 + (wave & 3) * 16;

    if (tid < 64) smin[tid] = 0x7f800000u;  // +inf bits
    __syncthreads();

    const _Float16* Gbase = Gn + ((size_t)b * P + q0 + lo) * C;
    float rmin = 3.0e38f;
    for (int p0 = (wave >> 2) * 16; p0 < P; p0 += 32) {
        const _Float16* Tbase = Tn + ((size_t)b * P + p0 + lo) * C;
        v8f acc = cosine_tile<C>(Tbase, Gbase, hi);
        #pragma unroll
        for (int r = 0; r < 8; ++r) {
            float cd = fmaxf(0.0f, (1.0f - acc[r]) * 0.5f);  // cdist, clamped >= 0
            rmin = fminf(rmin, cd);
        }
    }
    rmin = fminf(rmin, __shfl_xor(rmin, 16, 32));
    if (hi == 0) atomicMin(&smin[(wave & 3) * 16 + lo], __float_as_uint(rmin));
    __syncthreads();
    if (tid < 64) minc[(size_t)b * P + blockIdx.x * 64 + tid] = __uint_as_float(smin[tid]);
}

// ---------------------------------------------------------------------------
// Pass 2: per-column softmax denominator: sum_p exp((BIAS - rel)/SIGMA)
//         = sum_p exp(2*(1 - cdist/(minc+eps)))
// ---------------------------------------------------------------------------
template<int C, int P>
__global__ void colsum_kernel(const _Float16* __restrict__ Tn, const _Float16* __restrict__ Gn,
                              const float* __restrict__ minc, float* __restrict__ colsum) {
    __shared__ float ssum[64];
    const int tid = threadIdx.x;
    const int wave = tid >> 5, lane = tid & 31, lo = lane & 15, hi = lane >> 4;
    const int b = blockIdx.y;
    const int q0 = blockIdx.x * 64 + (wave & 3) * 16;

    if (tid < 64) ssum[tid] = 0.0f;
    __syncthreads();

    const _Float16* Gbase = Gn + ((size_t)b * P + q0 + lo) * C;
    const float invm = 1.0f / (minc[(size_t)b * P + q0 + lo] + EPS_DIV);
    float rsum = 0.0f;
    for (int p0 = (wave >> 2) * 16; p0 < P; p0 += 32) {
        const _Float16* Tbase = Tn + ((size_t)b * P + p0 + lo) * C;
        v8f acc = cosine_tile<C>(Tbase, Gbase, hi);
        #pragma unroll
        for (int r = 0; r < 8; ++r) {
            float cd = fmaxf(0.0f, (1.0f - acc[r]) * 0.5f);
            rsum += __expf(2.0f * (1.0f - cd * invm));
        }
    }
    rsum += __shfl_xor(rsum, 16, 32);
    if (hi == 0) atomicAdd(&ssum[(wave & 3) * 16 + lo], rsum);
    __syncthreads();
    if (tid < 64) colsum[(size_t)b * P + blockIdx.x * 64 + tid] = ssum[tid];
}

// ---------------------------------------------------------------------------
// Pass 3: per-row (per tar patch p) maximum of cs = e/colsum over all q.
// Block owns 64 rows; 8 waves = 4 p-subtiles x 2 q-phases; loop q.
// ---------------------------------------------------------------------------
template<int C, int P>
__global__ void rowmax_kernel(const _Float16* __restrict__ Tn, const _Float16* __restrict__ Gn,
                              const float* __restrict__ minc, const float* __restrict__ colsum,
                              float* __restrict__ rowmax) {
    __shared__ unsigned srm[64];
    const int tid = threadIdx.x;
    const int wave = tid >> 5, lane = tid & 31, lo = lane & 15, hi = lane >> 4;
    const int b = blockIdx.y;
    const int p0 = blockIdx.x * 64 + (wave & 3) * 16;

    if (tid < 64) srm[tid] = 0u;  // cs >= 0, so 0-bits is a valid identity
    __syncthreads();

    const _Float16* Tbase = Tn + ((size_t)b * P + p0 + lo) * C;
    float rmax[8];
    #pragma unroll
    for (int r = 0; r < 8; ++r) rmax[r] = 0.0f;

    for (int q0 = (wave >> 2) * 16; q0 < P; q0 += 32) {
        const _Float16* Gbase = Gn + ((size_t)b * P + q0 + lo) * C;
        v8f acc = cosine_tile<C>(Tbase, Gbase, hi);
        const float invm = 1.0f / (minc[(size_t)b * P + q0 + lo] + EPS_DIV);
        const float rcs  = 1.0f / colsum[(size_t)b * P + q0 + lo];
        #pragma unroll
        for (int r = 0; r < 8; ++r) {
            float cd = fmaxf(0.0f, (1.0f - acc[r]) * 0.5f);
            float cs = __expf(2.0f * (1.0f - cd * invm)) * rcs;
            rmax[r] = fmaxf(rmax[r], cs);
        }
    }
    // reduce max across the 16 lanes of each half-wave (columns of this tile)
    #pragma unroll
    for (int m = 1; m < 16; m <<= 1) {
        #pragma unroll
        for (int r = 0; r < 8; ++r)
            rmax[r] = fmaxf(rmax[r], __shfl_xor(rmax[r], m, 32));
    }
    // lane-holds-all: row index = (wave&3)*16 + 8*hi + r ; combine q-phases in LDS
    #pragma unroll
    for (int r = 0; r < 8; ++r)
        if (lo == r) atomicMax(&srm[(wave & 3) * 16 + 8 * hi + r], __float_as_uint(rmax[r]));
    __syncthreads();
    if (tid < 64) rowmax[(size_t)b * P + blockIdx.x * 64 + tid] = __uint_as_float(srm[tid]);
}

// ---------------------------------------------------------------------------
// Final: div_mrf = mean_p rowmax;  loss = sum_b -log(div+eps), layer42 weighted 2x
// ---------------------------------------------------------------------------
__global__ void finalize_kernel(const float* __restrict__ rowmaxA,
                                const float* __restrict__ rowmaxB,
                                float* __restrict__ out) {
    __shared__ float sbuf[256];
    const int tid = threadIdx.x;
    float loss = 0.0f;
    for (int combo = 0; combo < 8; ++combo) {
        const int layer = combo >> 2, b = combo & 3;
        const int P = layer ? PB : PA;
        const float* src = (layer ? rowmaxB : rowmaxA) + (size_t)b * P;
        float s = 0.0f;
        for (int i = tid; i < P; i += 256) s += src[i];
        sbuf[tid] = s;
        __syncthreads();
        for (int off = 128; off > 0; off >>= 1) {
            if (tid < off) sbuf[tid] += sbuf[tid + off];
            __syncthreads();
        }
        if (tid == 0) loss += -logf(sbuf[0] / (float)P + EPS_LOG) * (layer ? 2.0f : 1.0f);
        __syncthreads();
    }
    if (tid == 0) out[0] = loss;
}

// ---------------------------------------------------------------------------
extern "C" void kernel_launch(void* const* d_in, const int* in_sizes, int n_in,
                              void* d_out, int out_size, void* d_ws, size_t ws_size,
                              hipStream_t stream) {
    const float* gen32 = (const float*)d_in[0];
    const float* tar32 = (const float*)d_in[1];
    const float* gen42 = (const float*)d_in[2];
    const float* tar42 = (const float*)d_in[3];
    float* out = (float*)d_out;

    char* ws = (char*)d_ws;
    size_t off = 0;
    auto carve = [&](size_t bytes) -> void* {
        void* p = ws + off;
        off += (bytes + 255) & ~(size_t)255;
        return p;
    };
    _Float16* TnA = (_Float16*)carve((size_t)NB * PA * CA * 2);
    _Float16* GnA = (_Float16*)carve((size_t)NB * PA * CA * 2);
    _Float16* TnB = (_Float16*)carve((size_t)NB * PB * CB * 2);
    _Float16* GnB = (_Float16*)carve((size_t)NB * PB * CB * 2);
    float* mincA  = (float*)carve((size_t)NB * PA * 4);
    float* sumA   = (float*)carve((size_t)NB * PA * 4);
    float* rmaxA  = (float*)carve((size_t)NB * PA * 4);
    float* mincB  = (float*)carve((size_t)NB * PB * 4);
    float* sumB   = (float*)carve((size_t)NB * PB * 4);
    float* rmaxB  = (float*)carve((size_t)NB * PB * 4);
    if (off > ws_size) return;  // ~24.2 MB required

    prep_kernel<CA, PA><<<dim3(PA / 256, NB), 256, 0, stream>>>(gen32, tar32, GnA, TnA);
    prep_kernel<CB, PB><<<dim3(PB / 256, NB), 256, 0, stream>>>(gen42, tar42, GnB, TnB);

    colmin_kernel<CA, PA><<<dim3(PA / 64, NB), 256, 0, stream>>>(TnA, GnA, mincA);
    colmin_kernel<CB, PB><<<dim3(PB / 64, NB), 256, 0, stream>>>(TnB, GnB, mincB);

    colsum_kernel<CA, PA><<<dim3(PA / 64, NB), 256, 0, stream>>>(TnA, GnA, mincA, sumA);
    colsum_kernel<CB, PB><<<dim3(PB / 64, NB), 256, 0, stream>>>(TnB, GnB, mincB, sumB);

    rowmax_kernel<CA, PA><<<dim3(PA / 64, NB), 256, 0, stream>>>(TnA, GnA, mincA, sumA, rmaxA);
    rowmax_kernel<CB, PB><<<dim3(PB / 64, NB), 256, 0, stream>>>(TnB, GnB, mincB, sumB, rmaxB);

    finalize_kernel<<<1, 256, 0, stream>>>(rmaxA, rmaxB, out);
}
